// OptimizedDecoderLayer_16071767621715
// MI455X (gfx1250) — matrix operand order repaired and checked by reading, not compiled, not run on hardware
//
#include <hip/hip_runtime.h>
#include <hip/hip_bf16.h>

#define D_MODEL   2048
#define NUM_HEADS 16
#define HEAD_DIM  128
#define WINDOW    1024
#define BATCH     2
#define SEQ       2048
#define SCALE     0.08838834764831845f   // 1/sqrt(128)

typedef _Float16 h16;
typedef __attribute__((ext_vector_type(16))) _Float16 v16h;
typedef __attribute__((ext_vector_type(8)))  _Float16 v8h;
typedef __attribute__((ext_vector_type(8)))  float    v8f;

union V16 { v16h v; v8h h[2]; };

static __device__ __forceinline__ v8f wmma_f16(v16h a, v16h b, v8f c) {
  // D = A(16x32 f16) * B(32x16 f16) + C(16x16 f32)
  return __builtin_amdgcn_wmma_f32_16x16x32_f16(false, a, false, b, (short)0, c,
                                                false, false);
}

// ---------------------------------------------------------------------------
// fp32 -> fp16 conversion (grid-stride)
// ---------------------------------------------------------------------------
__global__ void k_f32_to_f16(const float* __restrict__ src,
                             h16* __restrict__ dst, int n) {
  int i = blockIdx.x * blockDim.x + threadIdx.x;
  int stride = gridDim.x * blockDim.x;
  for (; i < n; i += stride) dst[i] = (h16)src[i];
}

// ---------------------------------------------------------------------------
// QKV projection: out = X @ W^T + b, split heads.
// Register-blocked: one wave computes a 32(M) x 64(N) block
// (2 x 4 WMMA tiles, 8 v8f accumulators). Per 32-k step: 12 b128 loads feed
// 8 WMMAs (A reused 4x, B reused 2x).
// transposedStore==0: out layout [B,H,S,hd] (Q, K)
// transposedStore==1: out layout [B,H,hd,S] (V^T, contiguous PV B-operand)
// ---------------------------------------------------------------------------
__global__ void k_qkv_gemm(const h16* __restrict__ X, const h16* __restrict__ W,
                           const float* __restrict__ bias, h16* __restrict__ out,
                           int transposedStore) {
  const int lane = threadIdx.x & 31;
  const int wid  = threadIdx.x >> 5;
  const int b    = blockIdx.y;
  const int tile = blockIdx.x * 4 + wid;         // (S/32)*(D/64) = 2048 tiles
  const int s0   = (tile >> 5) << 5;             // 32-row block
  const int e0   = (tile & 31) << 6;             // 64-col block
  const int l15  = lane & 15;
  const int hi   = lane >> 4;

  const h16* arow0 = X + ((size_t)(b * SEQ + s0 +      l15)) * D_MODEL + hi * 16;
  const h16* arow1 = X + ((size_t)(b * SEQ + s0 + 16 + l15)) * D_MODEL + hi * 16;
  const h16* brow[4];
#pragma unroll
  for (int nt = 0; nt < 4; ++nt)
    brow[nt] = W + ((size_t)(e0 + nt * 16 + l15)) * D_MODEL + hi * 16;

  v8f acc[2][4];
#pragma unroll
  for (int mt = 0; mt < 2; ++mt)
#pragma unroll
    for (int nt = 0; nt < 4; ++nt) acc[mt][nt] = v8f{};

  for (int kb = 0; kb < D_MODEL; kb += 32) {
    V16 A[2], Bm[4];
    A[0].h[0] = *(const v8h*)(arow0 + kb);
    A[0].h[1] = *(const v8h*)(arow0 + kb + 8);
    A[1].h[0] = *(const v8h*)(arow1 + kb);
    A[1].h[1] = *(const v8h*)(arow1 + kb + 8);
#pragma unroll
    for (int nt = 0; nt < 4; ++nt) Bm[nt].v = *(const v16h*)(brow[nt] + kb);

    if (kb + 32 < D_MODEL) {                 // prefetch next A k-chunk
      __builtin_prefetch(arow0 + kb + 32, 0, 1);
      __builtin_prefetch(arow1 + kb + 32, 0, 1);
    }

#pragma unroll
    for (int mt = 0; mt < 2; ++mt)
#pragma unroll
      for (int nt = 0; nt < 4; ++nt)
        acc[mt][nt] = wmma_f16(A[mt].v, Bm[nt].v, acc[mt][nt]);
  }

  const int h = e0 >> 7;                     // whole 64-wide block in one head
#pragma unroll
  for (int nt = 0; nt < 4; ++nt) {
    const float bvv = bias[e0 + nt * 16 + l15];
    const int   d0  = (e0 & 127) + nt * 16;
    if (!transposedStore) {
      h16* obase = out + (size_t)(b * NUM_HEADS + h) * SEQ * HEAD_DIM;
#pragma unroll
      for (int mt = 0; mt < 2; ++mt)
#pragma unroll
        for (int r = 0; r < 8; ++r) {
          int s = s0 + mt * 16 + r + hi * 8;
          obase[(size_t)s * HEAD_DIM + d0 + l15] = (h16)(acc[mt][nt][r] + bvv);
        }
    } else {
      // [B,H,hd,S]: per-lane contiguous 8 halfs along s
      h16* obase = out + ((size_t)(b * NUM_HEADS + h) * HEAD_DIM + d0 + l15) * SEQ
                       + s0 + hi * 8;
#pragma unroll
      for (int mt = 0; mt < 2; ++mt) {
        v8h o;
#pragma unroll
        for (int r = 0; r < 8; ++r) o[r] = (h16)(acc[mt][nt][r] + bvv);
        *(v8h*)(obase + mt * 16) = o;
      }
    }
  }
}

// ---------------------------------------------------------------------------
// One 32-k step of windowed flash attention (MASK only in prologue iters).
// Scores computed transposed (rows=k, cols=q): the f32 C layout then matches
// the f16 A layout needed for P·V with zero cross-lane movement.
// ---------------------------------------------------------------------------
template <bool MASK>
static __device__ __forceinline__ void attn_step(
    int kb, int q0, int l15, int hi,
    const h16* __restrict__ Kh, const h16* __restrict__ Vh,
    const V16* __restrict__ qb, float& m_run, float& l_run, v8f* acc) {
  v8f sc[2];
#pragma unroll
  for (int t = 0; t < 2; ++t) {
    const int kt = kb + 16 * t;
    const h16* krow = Kh + (size_t)(kt + l15) * HEAD_DIM + hi * 16;
    v8f s = {};
#pragma unroll
    for (int c = 0; c < 4; ++c) {
      V16 A;
      A.h[0] = *(const v8h*)(krow + c * 32);
      A.h[1] = *(const v8h*)(krow + c * 32 + 8);
      s = wmma_f16(A.v, qb[c].v, s);
    }
    sc[t] = s;
  }

  const int qidx = q0 + l15;
  float tmax = -1e30f;
#pragma unroll
  for (int t = 0; t < 2; ++t) {
#pragma unroll
    for (int r = 0; r < 8; ++r) {
      float sv = sc[t][r] * SCALE;
      if (MASK) {
        const int kidx = kb + 16 * t + hi * 8 + r;
        if (qidx - kidx > WINDOW) sv = -1e30f;
      }
      sc[t][r] = sv;
      tmax = fmaxf(tmax, sv);
    }
  }
  tmax = fmaxf(tmax, __shfl_xor(tmax, 16, 32));   // combine k-halves per q
  const float newm  = fmaxf(m_run, tmax);
  const float alpha = __expf(m_run - newm);
  m_run = newm;

  float rsum = 0.f;
  V16 pA;
#pragma unroll
  for (int t = 0; t < 2; ++t) {
#pragma unroll
    for (int r = 0; r < 8; ++r) {
      const float p = __expf(sc[t][r] - newm);
      rsum += p;
      pA.v[t * 8 + r] = (h16)p;
    }
  }
  rsum += __shfl_xor(rsum, 16, 32);
  l_run = l_run * alpha + rsum;

#pragma unroll
  for (int i = 0; i < 8; ++i)
#pragma unroll
    for (int r = 0; r < 8; ++r) acc[i][r] *= alpha;

#pragma unroll
  for (int dt = 0; dt < 8; ++dt) {
    const h16* vrow = Vh + (size_t)(dt * 16 + l15) * SEQ + kb + hi * 8;
    V16 Bv;
    Bv.h[0] = *(const v8h*)vrow;
    Bv.h[1] = *(const v8h*)(vrow + 16);
    acc[dt] = wmma_f16(pA.v, Bv.v, acc[dt]);
  }
}

// ---------------------------------------------------------------------------
// Windowed flash attention. One wave per (b, h, 16-row q tile).
// Mask: allowed iff q - k <= WINDOW (future always allowed, per reference).
// ---------------------------------------------------------------------------
__global__ void k_attn(const h16* __restrict__ Q, const h16* __restrict__ K,
                       const h16* __restrict__ Vt, h16* __restrict__ O) {
  const int lane = threadIdx.x & 31;
  const int wid  = threadIdx.x >> 5;
  const int h    = blockIdx.y;
  const int b    = blockIdx.z;
  const int q0   = (blockIdx.x * 4 + wid) * 16;
  const int l15  = lane & 15;
  const int hi   = lane >> 4;

  const size_t headoff = (size_t)(b * NUM_HEADS + h) * SEQ * HEAD_DIM;
  const h16* Qh = Q  + headoff;
  const h16* Kh = K  + headoff;
  const h16* Vh = Vt + headoff;       // [hd, S]

  // Preload Q as B-operands for the 4 d-chunks (reused every k iteration)
  const h16* qrow = Qh + (size_t)(q0 + l15) * HEAD_DIM + hi * 16;
  V16 qb[4];
#pragma unroll
  for (int c = 0; c < 4; ++c) qb[c].v = *(const v16h*)(qrow + c * 32);

  float m_run = -1e30f, l_run = 0.f;
  v8f acc[8];
#pragma unroll
  for (int i = 0; i < 8; ++i) acc[i] = v8f{};

  int kstart = q0 - WINDOW;
  if (kstart < 0) kstart = 0;
  kstart &= ~31;
  const int kmask_end = q0 + 15 - WINDOW;   // kb < this may contain masked elems

  int kb = kstart;
  for (; kb < SEQ && kb < kmask_end; kb += 32)
    attn_step<true>(kb, q0, l15, hi, Kh, Vh, qb, m_run, l_run, acc);
  for (; kb < SEQ; kb += 32)
    attn_step<false>(kb, q0, l15, hi, Kh, Vh, qb, m_run, l_run, acc);

  // ---- normalize + store O [B,H,S,hd] ----
  const float inv = 1.f / l_run;
  h16* obase = O + headoff;
#pragma unroll
  for (int dt = 0; dt < 8; ++dt) {
#pragma unroll
    for (int r = 0; r < 8; ++r) {
      const int q = q0 + r + hi * 8;
      obase[(size_t)q * HEAD_DIM + dt * 16 + l15] = (h16)(acc[dt][r] * inv);
    }
  }
}

// ---------------------------------------------------------------------------
// Output projection: out(f32) = O @ Wo^T + bo ; O stored [B,H,S,hd].
// Same 32x64 register blocking as the QKV GEMM.
// ---------------------------------------------------------------------------
__global__ void k_oproj(const h16* __restrict__ Oh, const h16* __restrict__ Wo,
                        const float* __restrict__ bo, float* __restrict__ out) {
  const int lane = threadIdx.x & 31;
  const int wid  = threadIdx.x >> 5;
  const int b    = blockIdx.y;
  const int tile = blockIdx.x * 4 + wid;
  const int s0   = (tile >> 5) << 5;
  const int e0   = (tile & 31) << 6;
  const int l15  = lane & 15;
  const int hi   = lane >> 4;

  const h16* brow[4];
#pragma unroll
  for (int nt = 0; nt < 4; ++nt)
    brow[nt] = Wo + ((size_t)(e0 + nt * 16 + l15)) * D_MODEL + hi * 16;

  v8f acc[2][4];
#pragma unroll
  for (int mt = 0; mt < 2; ++mt)
#pragma unroll
    for (int nt = 0; nt < 4; ++nt) acc[mt][nt] = v8f{};

  for (int kb = 0; kb < D_MODEL; kb += 32) {
    const int hh   = kb >> 7;    // head supplying this k chunk
    const int doff = kb & 127;
    const h16* ab = Oh + (size_t)(b * NUM_HEADS + hh) * SEQ * HEAD_DIM
                       + doff + hi * 16;
    V16 A[2], Bm[4];
#pragma unroll
    for (int mt = 0; mt < 2; ++mt) {
      const h16* arow = ab + (size_t)(s0 + mt * 16 + l15) * HEAD_DIM;
      A[mt].h[0] = *(const v8h*)(arow);
      A[mt].h[1] = *(const v8h*)(arow + 8);
    }
#pragma unroll
    for (int nt = 0; nt < 4; ++nt) Bm[nt].v = *(const v16h*)(brow[nt] + kb);

#pragma unroll
    for (int mt = 0; mt < 2; ++mt)
#pragma unroll
      for (int nt = 0; nt < 4; ++nt)
        acc[mt][nt] = wmma_f16(A[mt].v, Bm[nt].v, acc[mt][nt]);
  }

#pragma unroll
  for (int nt = 0; nt < 4; ++nt) {
    const float bvv = bo[e0 + nt * 16 + l15];
#pragma unroll
    for (int mt = 0; mt < 2; ++mt)
#pragma unroll
      for (int r = 0; r < 8; ++r) {
        const int s = s0 + mt * 16 + r + hi * 8;
        out[((size_t)b * SEQ + s) * D_MODEL + e0 + nt * 16 + l15] =
            acc[mt][nt][r] + bvv;
      }
  }
}

// ---------------------------------------------------------------------------
extern "C" void kernel_launch(void* const* d_in, const int* in_sizes, int n_in,
                              void* d_out, int out_size, void* d_ws, size_t ws_size,
                              hipStream_t stream) {
  (void)in_sizes; (void)n_in; (void)out_size; (void)ws_size;

  const float* hs = (const float*)d_in[0];
  const float* Wq = (const float*)d_in[1];
  const float* bq = (const float*)d_in[2];
  const float* Wk = (const float*)d_in[3];
  const float* bk = (const float*)d_in[4];
  const float* Wv = (const float*)d_in[5];
  const float* bv = (const float*)d_in[6];
  const float* Wo = (const float*)d_in[7];
  const float* bo = (const float*)d_in[8];

  const size_t nX = (size_t)BATCH * SEQ * D_MODEL;   // 8,388,608
  const size_t nW = (size_t)D_MODEL * D_MODEL;       // 4,194,304

  h16* p   = (h16*)d_ws;
  h16* hX  = p; p += nX;
  h16* hWq = p; p += nW;
  h16* hWk = p; p += nW;
  h16* hWv = p; p += nW;
  h16* hWo = p; p += nW;
  h16* hQ  = p; p += nX;
  h16* hK  = p; p += nX;
  h16* hVt = p; p += nX;
  h16* hO  = p; p += nX;
  // total workspace: (5*nX + 4*nW)*2B = 112 MiB

  k_f32_to_f16<<<4096, 256, 0, stream>>>(hs, hX, (int)nX);
  k_f32_to_f16<<<2048, 256, 0, stream>>>(Wq, hWq, (int)nW);
  k_f32_to_f16<<<2048, 256, 0, stream>>>(Wk, hWk, (int)nW);
  k_f32_to_f16<<<2048, 256, 0, stream>>>(Wv, hWv, (int)nW);
  k_f32_to_f16<<<2048, 256, 0, stream>>>(Wo, hWo, (int)nW);

  // 2048 (32x64) tiles per batch, 4 waves/block -> 512 blocks
  dim3 gg(512, BATCH);
  k_qkv_gemm<<<gg, 128, 0, stream>>>(hX, hWq, bq, hQ, 0);
  k_qkv_gemm<<<gg, 128, 0, stream>>>(hX, hWk, bk, hK, 0);
  k_qkv_gemm<<<gg, 128, 0, stream>>>(hX, hWv, bv, hVt, 1);  // V transposed

  k_attn<<<dim3(32, NUM_HEADS, BATCH), 128, 0, stream>>>(hQ, hK, hVt, hO);

  k_oproj<<<gg, 128, 0, stream>>>(hO, hWo, bo, (float*)d_out);
}
